// MaximumMeanDiscrepancyLoss_32633161515875
// MI455X (gfx1250) — matrix operand order repaired and checked by reading, not compiled, not run on hardware
//
#include <hip/hip_runtime.h>
#include <hip/hip_bf16.h>

// ---------------------------------------------------------------------------
// MMD^2 (RBF, sigma=1) on x,y: 4096x1024 f32 each.
// Z = [x;y] (8192x1024). One symmetric bf16-WMMA GEMM Z*Z^T with fused
// exp + weighted-sum epilogue. Only elements j > i are accumulated (weight
// 2x), which reproduces the full ordered sum and drops the diagonal.
// Block tile: 256 rows x 128 cols; 8 wave32s in a 4x2 grid, 64x64 per wave
// (16 WMMAs per 16 ds_load_b128 per k-step -> compute-bound vs LDS).
// ---------------------------------------------------------------------------

typedef __attribute__((ext_vector_type(16))) __bf16 v16bf;
typedef __attribute__((ext_vector_type(8)))  __bf16 v8bf;
typedef __attribute__((ext_vector_type(4)))  __bf16 v4bf;
typedef __attribute__((ext_vector_type(8)))  float  v8f;

#define N_PTS 4096
#define DIM   1024
#define NTOT  8192          // 2 * N_PTS
#define BMR   256           // block tile rows
#define BN    128           // block tile cols
#define BK    64            // k-chunk (elements)
#define LDA   72            // padded LDS row stride (elements); 144B = 9*16B
#define NBI   32            // 8192 / 256 (row blocks)
#define NBJ   64            // 8192 / 128 (col blocks)
#define NPART (NBI * NBJ)   // 2048 partials

// ---------------- kernel 1: exact f32 row norms of Z --------------------
__global__ __launch_bounds__(256)
void row_norms_kernel(const float* __restrict__ x, const float* __restrict__ y,
                      float* __restrict__ norms) {
  const int row = blockIdx.x;
  const float* src = (row < N_PTS) ? (x + (size_t)row * DIM)
                                   : (y + (size_t)(row - N_PTS) * DIM);
  const int t = threadIdx.x;
  const float4 v = ((const float4*)src)[t];      // 256 threads * 4 = 1024
  float s = v.x * v.x + v.y * v.y + v.z * v.z + v.w * v.w;
  __shared__ float red[256];
  red[t] = s;
  __syncthreads();
  for (int off = 128; off > 0; off >>= 1) {
    if (t < off) red[t] += red[t + off];
    __syncthreads();
  }
  if (t == 0) norms[row] = red[0];
}

// ------------- LDS fragment loaders (ISA 16-bit A/B layouts) ------------
// A 16x32 (bf16): lanes 0-15 rows M=0..15, lanes 16-31 same rows, k += 8.
// Per lane: k = {kb..kb+7, kb+16..kb+23}, kb = kk + (lane>=16 ? 8 : 0).
__device__ __forceinline__ v16bf load_fragA(const __bf16* base, int row, int kbase) {
  const __bf16* p = base + row * LDA + kbase;
  v8bf lo = *(const v8bf*)p;          // k = kb .. kb+7      (16B, aligned)
  v8bf hi = *(const v8bf*)(p + 16);   // k = kb+16 .. kb+23
  v16bf r;
#pragma unroll
  for (int e = 0; e < 8; ++e) { r[e] = lo[e]; r[8 + e] = hi[e]; }
  return r;
}

// B 32x16 (bf16): same k order as A: k = {kb..kb+7, kb+16..kb+23}, kb = kk + (lane>=16 ? 8 : 0).
// We store B as rows-of-Z (i.e. B^T).
__device__ __forceinline__ v16bf load_fragB(const __bf16* base, int col, int kbase) {
  const __bf16* p = base + col * LDA + kbase;
  v8bf lo = *(const v8bf*)p;
  v8bf hi = *(const v8bf*)(p + 16);
  v16bf r;
#pragma unroll
  for (int e = 0; e < 8; ++e) { r[e] = lo[e]; r[8 + e] = hi[e]; }
  return r;
}

// ---------------- kernel 2: fused tile GEMM + exp + reduce --------------
__global__ __launch_bounds__(256)
void mmd_tile_kernel(const float* __restrict__ x, const float* __restrict__ y,
                     const float* __restrict__ norms, float* __restrict__ partials) {
  const int bi = blockIdx.y;            // row block (256 rows)
  const int bj = blockIdx.x;            // col block (128 cols)
  const int pidx = bi * NBJ + bj;
  const int rowBase = bi * BMR;
  const int colBase = bj * BN;
  // Skip blocks with no j > i elements (entirely at/below the diagonal).
  if (colBase + BN - 1 <= rowBase) {
    if (threadIdx.x == 0) partials[pidx] = 0.0f;
    return;
  }

  __shared__ __bf16 As[BMR][LDA];       // 256 x 72 bf16 = 36 KB
  __shared__ __bf16 Bs[BN][LDA];        // 128 x 72 bf16 = 18 KB
  __shared__ float  wred[8];

  const int tid   = threadIdx.x;
  const int lane  = tid & 31;
  const int wave  = tid >> 5;           // 0..7
  const int waveM = wave >> 1;          // 0..3 -> 64-row slab
  const int waveN = wave & 1;           // 0..1 -> 64-col slab

  // Global-load assignment:
  //   A: thread t loads full 64-wide chunk of row (rowBase + t)     (16 float4)
  //   B: thread t loads half chunk of row (colBase + t/2)           ( 8 float4)
  const int ra = rowBase + tid;
  const int rbRow = tid >> 1;
  const int rb = colBase + rbRow;
  const int bcol = (tid & 1) * 32;
  const float* rowA = (ra < N_PTS) ? x + (size_t)ra * DIM : y + (size_t)(ra - N_PTS) * DIM;
  const float* rowB = (rb < N_PTS) ? x + (size_t)rb * DIM : y + (size_t)(rb - N_PTS) * DIM;

  v8f acc[4][4];
#pragma unroll
  for (int mt = 0; mt < 4; ++mt)
#pragma unroll
    for (int nt = 0; nt < 4; ++nt)
#pragma unroll
      for (int e = 0; e < 8; ++e) acc[mt][nt][e] = 0.0f;

  const int jlane = lane & 15;
  const int hiK8  = (lane >> 4) * 8;    // A-frag k shift for lanes 16..31
  const int hiK16 = (lane >> 4) * 16;   // unused; B frags also shift by hiK8

  for (int k0 = 0; k0 < DIM; k0 += BK) {
    __syncthreads();                    // previous chunk's frags consumed
    const float4* a4 = (const float4*)(rowA + k0);
    const float4* b4 = (const float4*)(rowB + k0 + bcol);
#pragma unroll
    for (int q = 0; q < 16; ++q) {      // A: 64 floats of one row
      const float4 av = a4[q];
      v4bf ac;
      ac[0] = (__bf16)av.x; ac[1] = (__bf16)av.y; ac[2] = (__bf16)av.z; ac[3] = (__bf16)av.w;
      *(v4bf*)&As[tid][4 * q] = ac;
    }
#pragma unroll
    for (int q = 0; q < 8; ++q) {       // B: 32 floats of half a row
      const float4 bv = b4[q];
      v4bf bc;
      bc[0] = (__bf16)bv.x; bc[1] = (__bf16)bv.y; bc[2] = (__bf16)bv.z; bc[3] = (__bf16)bv.w;
      *(v4bf*)&Bs[rbRow][bcol + 4 * q] = bc;
    }
    __syncthreads();

#pragma unroll
    for (int kk = 0; kk < BK; kk += 32) {
      v16bf a[4], b[4];
#pragma unroll
      for (int mt = 0; mt < 4; ++mt)
        a[mt] = load_fragA(&As[0][0], waveM * 64 + mt * 16 + jlane, kk + hiK8);
#pragma unroll
      for (int nt = 0; nt < 4; ++nt)
        b[nt] = load_fragB(&Bs[0][0], waveN * 64 + nt * 16 + jlane, kk + hiK8);
#pragma unroll
      for (int mt = 0; mt < 4; ++mt)
#pragma unroll
        for (int nt = 0; nt < 4; ++nt)
          acc[mt][nt] = __builtin_amdgcn_wmma_f32_16x16x32_bf16(
              false, a[mt], false, b[nt], (short)0, acc[mt][nt], false, false);
    }
  }

  // ---- epilogue: d = aa + bb - 2*ab, exp, j>i weighting, reduce ----
  // Only j>i contributes, with weight 2*base (base symmetric, k symmetric):
  //   same-set base: 1/(n(n-1)), cross-set base: -1/(n*m). Diagonal excluded.
  const float cxx2 = 2.0f / ((float)N_PTS * (float)(N_PTS - 1));
  const float cxy2 = -2.0f / ((float)N_PTS * (float)N_PTS);

  float tsum = 0.0f;
#pragma unroll
  for (int mt = 0; mt < 4; ++mt) {
    const int ibase = rowBase + waveM * 64 + mt * 16 + hiK8;      // C/D: VGPR r -> M
    float aa[8];
#pragma unroll
    for (int r = 0; r < 8; ++r) aa[r] = norms[ibase + r];
#pragma unroll
    for (int nt = 0; nt < 4; ++nt) {
      const int j  = colBase + waveN * 64 + nt * 16 + jlane;      // C/D: lane -> N
      const float bb = norms[j];
#pragma unroll
      for (int r = 0; r < 8; ++r) {
        const int i = ibase + r;
        float d = aa[r] + bb - 2.0f * acc[mt][nt][r];
        d = fmaxf(d, 0.0f);
        const float kv = __expf(-0.5f * d);                       // sigma = 1
        const bool ix = i < N_PTS, jx = j < N_PTS;
        const float base = (ix == jx) ? cxx2 : cxy2;
        const float w = (j > i) ? base : 0.0f;
        tsum += w * kv;
      }
    }
  }

  // wave32 reduce, then cross-wave via LDS
#pragma unroll
  for (int m = 16; m >= 1; m >>= 1) tsum += __shfl_xor(tsum, m, 32);
  if (lane == 0) wred[wave] = tsum;
  __syncthreads();
  if (tid == 0) {
    float s = 0.0f;
#pragma unroll
    for (int wv = 0; wv < 8; ++wv) s += wred[wv];
    partials[pidx] = s;
  }
}

// ---------------- kernel 3: deterministic final reduction ----------------
__global__ __launch_bounds__(256)
void final_reduce_kernel(const float* __restrict__ partials, float* __restrict__ out) {
  __shared__ float red[256];
  const int t = threadIdx.x;
  float s = 0.0f;
  for (int i = t; i < NPART; i += 256) s += partials[i];          // fixed order
  red[t] = s;
  __syncthreads();
  for (int off = 128; off > 0; off >>= 1) {
    if (t < off) red[t] += red[t + off];
    __syncthreads();
  }
  if (t == 0) out[0] = red[0];
}

extern "C" void kernel_launch(void* const* d_in, const int* in_sizes, int n_in,
                              void* d_out, int out_size, void* d_ws, size_t ws_size,
                              hipStream_t stream) {
  (void)in_sizes; (void)n_in; (void)out_size; (void)ws_size;
  const float* x = (const float*)d_in[0];
  const float* y = (const float*)d_in[1];
  float* out      = (float*)d_out;
  float* norms    = (float*)d_ws;        // 8192 floats
  float* partials = norms + NTOT;        // 2048 floats (total ws use: 40 KB)

  row_norms_kernel<<<NTOT, 256, 0, stream>>>(x, y, norms);
  dim3 grid(NBJ, NBI);
  mmd_tile_kernel<<<grid, 256, 0, stream>>>(x, y, norms, partials);
  final_reduce_kernel<<<1, 256, 0, stream>>>(partials, out);
}
